// GGNN_40132174414161
// MI455X (gfx1250) — matrix operand order repaired and checked by reading, not compiled, not run on hardware
//
#include <hip/hip_runtime.h>
#include <hip/hip_bf16.h>

// ---------------------------------------------------------------------------
// GGNN on MI455X (gfx1250), compile-only target.
// GEMMs via v_wmma_f32_16x16x32_bf16 (wave32), scatter via f32 atomics.
// ---------------------------------------------------------------------------

typedef __attribute__((ext_vector_type(16))) __bf16 bf16x16;
typedef __attribute__((ext_vector_type(8)))  __bf16 bf16x8;
typedef __attribute__((ext_vector_type(8)))  float  v8f;

__device__ __forceinline__ unsigned short f32_to_bf16(float f) {
    unsigned int u = __float_as_uint(f);
    u += 0x7fffu + ((u >> 16) & 1u);          // round-to-nearest-even
    return (unsigned short)(u >> 16);
}

// ------------------------- elementwise helpers -----------------------------

__global__ __launch_bounds__(256) void copy_f32_kernel(const float* __restrict__ in,
                                                       float* __restrict__ out, int n) {
    int i = blockIdx.x * 256 + threadIdx.x;
    if (i < n) out[i] = in[i];
}

__global__ __launch_bounds__(256) void zero_f32_kernel(float* __restrict__ p, int n) {
    int i = blockIdx.x * 256 + threadIdx.x;
    if (i < n) p[i] = 0.0f;
}

__global__ __launch_bounds__(256) void f2bf_kernel(const float* __restrict__ in,
                                                   unsigned short* __restrict__ out, int n) {
    int i = blockIdx.x * 256 + threadIdx.x;
    if (i < n) out[i] = f32_to_bf16(in[i]);
}

// ------------------------- WMMA bf16 GEMM ----------------------------------
// C[M x Nout] = A[M x 128] (bf16, row-major) * W[Nout x 128]^T (bf16) + bias.
// One wave per 16-row M-tile; A k-fragments cached in registers, sweep N-tiles.
__global__ __launch_bounds__(256) void wmma_gemm_bf16(const unsigned short* __restrict__ A,
                                                      const unsigned short* __restrict__ W,
                                                      const float* __restrict__ bias,
                                                      float* __restrict__ C,
                                                      int M, int Nout) {
    const int K = 128;
    const int lane = threadIdx.x & 31;
    const int wid  = (blockIdx.x * blockDim.x + threadIdx.x) >> 5;   // global wave id
    const int mtiles = M >> 4;
    if (wid >= mtiles) return;

    const int r  = lane & 15;        // row within tile (A) / column (B) / N (C)
    const int hi = lane >> 4;        // lane-half selector

    // A fragments: lane<16 holds K {k0+0..7, k0+16..23}; lane>=16 holds {+8..15, +24..31}
    const unsigned short* arow = A + (size_t)(wid * 16 + r) * K + hi * 8;
    bf16x16 afrag[4];
#pragma unroll
    for (int kk = 0; kk < 4; ++kk) {
        bf16x8 lo = *(const bf16x8*)(arow + kk * 32);
        bf16x8 hi8 = *(const bf16x8*)(arow + kk * 32 + 16);
        afrag[kk] = __builtin_shufflevector(lo, hi8,
                                            0, 1, 2, 3, 4, 5, 6, 7,
                                            8, 9, 10, 11, 12, 13, 14, 15);
    }

    const int ntiles = Nout >> 4;
    for (int tn = 0; tn < ntiles; ++tn) {
        v8f acc = {};
        // B fragment: column N = r (row tn*16+r of W), K = kk*32 + {hi*8 + [0..7], 16 + hi*8 + [0..7]}
        const unsigned short* brow = W + (size_t)(tn * 16 + r) * K + hi * 8;
#pragma unroll
        for (int kk = 0; kk < 4; ++kk) {
            bf16x8 blo = *(const bf16x8*)(brow + kk * 32);
            bf16x8 bhi = *(const bf16x8*)(brow + kk * 32 + 16);
            bf16x16 bfrag = __builtin_shufflevector(blo, bhi,
                                                    0, 1, 2, 3, 4, 5, 6, 7,
                                                    8, 9, 10, 11, 12, 13, 14, 15);
            acc = __builtin_amdgcn_wmma_f32_16x16x32_bf16(
                false, afrag[kk], false, bfrag, (short)0, acc, false, false);
        }
        const int col = tn * 16 + r;
        const float bb = bias[col];
        const size_t rowbase = (size_t)(wid * 16 + hi * 8);
#pragma unroll
        for (int v = 0; v < 8; ++v)              // acc[v]: row = rowbase+v, col
            C[(rowbase + v) * (size_t)Nout + col] = acc[v] + bb;
    }
}

// ------------------------- edge scatter-sum --------------------------------
// a[dst[e], :] += m[src[e], :]   (H = 128 channels, one thread per element)
__global__ __launch_bounds__(256) void scatter_add_kernel(const float* __restrict__ m,
                                                          const int* __restrict__ src,
                                                          const int* __restrict__ dst,
                                                          float* __restrict__ a, int E) {
    long long idx = (long long)blockIdx.x * 256 + threadIdx.x;
    if (idx >= (long long)E * 128) return;
    int e = (int)(idx >> 7);
    int c = (int)(idx & 127);
    float v = m[(size_t)src[e] * 128 + c];
    atomicAdd(&a[(size_t)dst[e] * 128 + c], v);
}

// ------------------------- GRU gates ---------------------------------------
// gi/gh are [N x 384] with column blocks (r, z, n); h updated in place.
__global__ __launch_bounds__(256) void gru_kernel(const float* __restrict__ gi,
                                                  const float* __restrict__ gh,
                                                  float* __restrict__ h, int nh) {
    int idx = blockIdx.x * 256 + threadIdx.x;
    if (idx >= nh) return;
    int n = idx >> 7;
    int c = idx & 127;
    size_t base = (size_t)n * 384;
    float ir = gi[base + c],       iz = gi[base + 128 + c], inn = gi[base + 256 + c];
    float hr = gh[base + c],       hz = gh[base + 128 + c], hn  = gh[base + 256 + c];
    float r = 1.0f / (1.0f + __expf(-(ir + hr)));
    float z = 1.0f / (1.0f + __expf(-(iz + hz)));
    float nn = tanhf(inn + r * hn);
    float hv = h[idx];
    h[idx] = (1.0f - z) * nn + z * hv;
}

// ------------------------- pair feature gather -----------------------------
// feat[p, c] = bf16( h[s[p], c] * h[d[p], c] )
__global__ __launch_bounds__(256) void pair_feat_kernel(const float* __restrict__ h,
                                                        const int* __restrict__ s,
                                                        const int* __restrict__ d,
                                                        unsigned short* __restrict__ feat,
                                                        int P) {
    int idx = blockIdx.x * 256 + threadIdx.x;
    if (idx >= P * 128) return;
    int p = idx >> 7;
    int c = idx & 127;
    float v = h[(size_t)s[p] * 128 + c] * h[(size_t)d[p] * 128 + c];
    feat[idx] = f32_to_bf16(v);
}

// ------------------------- predictor tail ----------------------------------
// out[p] = b2 + sum_j W2[j] * leaky_relu_0.2(h1[p, j]),  j < 64
__global__ __launch_bounds__(256) void pred_tail_kernel(const float* __restrict__ h1,
                                                        const float* __restrict__ W2,
                                                        const float* __restrict__ b2,
                                                        float* __restrict__ out, int P) {
    int p = blockIdx.x * 256 + threadIdx.x;
    if (p >= P) return;
    const float* row = h1 + (size_t)p * 64;
    float acc = b2[0];
#pragma unroll 8
    for (int j = 0; j < 64; ++j) {
        float v = row[j];
        v = (v >= 0.0f) ? v : 0.2f * v;
        acc += W2[j] * v;
    }
    out[p] = acc;
}

// ---------------------------------------------------------------------------

extern "C" void kernel_launch(void* const* d_in, const int* in_sizes, int n_in,
                              void* d_out, int out_size, void* d_ws, size_t ws_size,
                              hipStream_t stream) {
    (void)in_sizes; (void)n_in; (void)out_size; (void)ws_size;
    const int N = 100000, E = 1600000, P = 200000, H = 128;
    const size_t NH = (size_t)N * H;

    const float* x       = (const float*)d_in[0];
    const int*   src     = (const int*)d_in[1];
    const int*   dst     = (const int*)d_in[2];
    const int*   pos_src = (const int*)d_in[3];
    const int*   pos_dst = (const int*)d_in[4];
    const int*   neg_src = (const int*)d_in[5];
    const int*   neg_dst = (const int*)d_in[6];
    const float* W_e     = (const float*)d_in[7];
    const float* b_e     = (const float*)d_in[8];
    const float* W_ih    = (const float*)d_in[9];
    const float* b_ih    = (const float*)d_in[10];
    const float* W_hh    = (const float*)d_in[11];
    const float* b_hh    = (const float*)d_in[12];
    const float* W1      = (const float*)d_in[13];
    const float* b1      = (const float*)d_in[14];
    const float* W2      = (const float*)d_in[15];
    const float* b2      = (const float*)d_in[16];
    float* out = (float*)d_out;

    // workspace layout (all offsets multiples of 128 elements -> aligned)
    float* h   = (float*)d_ws;                 // N*H f32
    float* aAg = h + NH;                       // N*H f32
    float* gi  = aAg + NH;                     // N*3H f32
    float* gh  = gi + 3 * NH;                  // N*3H f32
    unsigned short* hbf  = (unsigned short*)(gh + 3 * NH);  // N*H bf16
    unsigned short* abf  = hbf + NH;                        // N*H bf16
    unsigned short* web  = abf + NH;                        // 128*128
    unsigned short* wihb = web + H * H;                     // 384*128
    unsigned short* whhb = wihb + 3 * H * H;                // 384*128
    unsigned short* w1b  = whhb + 3 * H * H;                // 64*128
    float* m = gi;                              // alias: dead before gi written
    unsigned short* featbf = (unsigned short*)gi;  // alias after step loop
    float* h1 = gh;                                // alias after step loop

    auto blocks = [](size_t n) { return dim3((unsigned)((n + 255) / 256)); };

    // one-time weight conversions
    f2bf_kernel<<<blocks(H * H), 256, 0, stream>>>(W_e, web, H * H);
    f2bf_kernel<<<blocks(3 * H * H), 256, 0, stream>>>(W_ih, wihb, 3 * H * H);
    f2bf_kernel<<<blocks(3 * H * H), 256, 0, stream>>>(W_hh, whhb, 3 * H * H);
    f2bf_kernel<<<blocks(64 * H), 256, 0, stream>>>(W1, w1b, 64 * H);
    copy_f32_kernel<<<blocks(NH), 256, 0, stream>>>(x, h, (int)NH);

    const int wavesPerBlock = 8;                               // 256 threads
    dim3 gemmGridN((N / 16 + wavesPerBlock - 1) / wavesPerBlock);
    dim3 gemmGridP((P / 16 + wavesPerBlock - 1) / wavesPerBlock);

    for (int step = 0; step < 3; ++step) {
        f2bf_kernel<<<blocks(NH), 256, 0, stream>>>(h, hbf, (int)NH);
        // m = h @ W_e^T + b_e
        wmma_gemm_bf16<<<gemmGridN, 256, 0, stream>>>(hbf, web, b_e, m, N, H);
        // a = segment_sum(m[src], dst)
        zero_f32_kernel<<<blocks(NH), 256, 0, stream>>>(aAg, (int)NH);
        scatter_add_kernel<<<blocks((size_t)E * H), 256, 0, stream>>>(m, src, dst, aAg, E);
        f2bf_kernel<<<blocks(NH), 256, 0, stream>>>(aAg, abf, (int)NH);
        // gh = h @ W_hh^T + b_hh ; gi = a @ W_ih^T + b_ih  (gi overwrites dead m)
        wmma_gemm_bf16<<<gemmGridN, 256, 0, stream>>>(hbf, whhb, b_hh, gh, N, 3 * H);
        wmma_gemm_bf16<<<gemmGridN, 256, 0, stream>>>(abf, wihb, b_ih, gi, N, 3 * H);
        // h = GRU(a, h)
        gru_kernel<<<blocks(NH), 256, 0, stream>>>(gi, gh, h, (int)NH);
    }

    // predictor: pos pairs
    pair_feat_kernel<<<blocks((size_t)P * H), 256, 0, stream>>>(h, pos_src, pos_dst, featbf, P);
    wmma_gemm_bf16<<<gemmGridP, 256, 0, stream>>>(featbf, w1b, b1, h1, P, 64);
    pred_tail_kernel<<<blocks(P), 256, 0, stream>>>(h1, W2, b2, out, P);
    // predictor: neg pairs
    pair_feat_kernel<<<blocks((size_t)P * H), 256, 0, stream>>>(h, neg_src, neg_dst, featbf, P);
    wmma_gemm_bf16<<<gemmGridP, 256, 0, stream>>>(featbf, w1b, b1, h1, P, 64);
    pred_tail_kernel<<<blocks(P), 256, 0, stream>>>(h1, W2, b2, out + P, P);
    // final hidden state
    copy_f32_kernel<<<blocks(NH), 256, 0, stream>>>(h, out + 2 * (size_t)P, (int)NH);
}
